// EncoDecLSTM_40097814676016
// MI455X (gfx1250) — matrix operand order repaired and checked by reading, not compiled, not run on hardware
//
#include <hip/hip_runtime.h>

typedef _Float16 half_t;
typedef __attribute__((ext_vector_type(16))) _Float16 v16h;
typedef __attribute__((ext_vector_type(8)))  _Float16 v8h;
typedef __attribute__((ext_vector_type(8)))  float    v8f;
typedef __attribute__((ext_vector_type(4)))  unsigned int v4u;

#define B_SZ  256
#define T_LEN 512
#define F_DIM 128
#define U_DIM 1024

// ---------------- CDNA5 async global->LDS staging (guarded) ----------------
#if defined(__has_builtin)
#if __has_builtin(__builtin_amdgcn_global_load_async_to_lds_b128)
#define HAS_ASYNC_LDS 1
#endif
#endif
#ifndef HAS_ASYNC_LDS
#define HAS_ASYNC_LDS 0
#endif

// The builtin's params are pointers to a 16-byte int vector (per hipcc
// diagnostic: "int __vector(4) __device__ *"), in global / shared AS.
typedef int b128_t __attribute__((vector_size(16)));
typedef __attribute__((address_space(1))) b128_t* g128p;
typedef __attribute__((address_space(3))) b128_t* l128p;
typedef __attribute__((address_space(3))) void*   as3vp;

// Copy 8 halfs (16B) global -> LDS. Async DMA if available, else via VGPRs.
__device__ __forceinline__ void copy16_g2l(const half_t* __restrict__ g, half_t* l) {
#if HAS_ASYNC_LDS
  __builtin_amdgcn_global_load_async_to_lds_b128(
      (g128p)(unsigned long long)g, (l128p)(as3vp)l, 0, 0);
#else
  *(v4u*)l = *(const v4u*)g;
#endif
}

__device__ __forceinline__ void stage_fence() {
#if HAS_ASYNC_LDS
#if __has_builtin(__builtin_amdgcn_s_wait_asynccnt)
  __builtin_amdgcn_s_wait_asynccnt(0);
#else
  asm volatile("s_wait_asynccnt 0x0" ::: "memory");
#endif
#endif
  __syncthreads();
}

__device__ __forceinline__ v16h cat16(v8h lo, v8h hi) {
  v16h r;
#pragma unroll
  for (int i = 0; i < 8; ++i) { r[i] = lo[i]; r[i + 8] = hi[i]; }
  return r;
}

__device__ __forceinline__ float sigmoidf_(float x) {
  return 1.0f / (1.0f + __expf(-x));
}
__device__ __forceinline__ float tanhf_(float x) {
  return 1.0f - 2.0f / (1.0f + __expf(2.0f * x));  // overflow-safe
}

// ---------------------------------------------------------------------------
// Fused LSTM step: Z[B,4U] = A @ W^T + b, then gate nonlinearity.
// WT is COLUMN-major (k contiguous per output column): WT[col][k], col in [0,4U).
//   encoder: A = [x_t | h_prev], K = 1152;  decoder: A = h_prev, K = 1024.
// Grid (U/32, B/64), block 128 (4 waves). Each WG: 64x32 tile of all 4 gates.
// ---------------------------------------------------------------------------
template <bool IS_ENC>
__global__ __launch_bounds__(128) void lstm_step_kernel(
    const half_t* __restrict__ xh, int t,
    const half_t* __restrict__ hprev, float* __restrict__ cbuf,
    const half_t* __restrict__ WT, const float* __restrict__ bias,
    half_t* __restrict__ hnext) {
  constexpr int K  = IS_ENC ? (F_DIM + U_DIM) : U_DIM;
  constexpr int KT = K / 32;

  __shared__ __align__(16) half_t sA[2][64 * 32];     // row-major [row][k]
  __shared__ __align__(16) half_t sB[2][4][32 * 40];  // col-major [n][k], pad 40

  const int tid   = threadIdx.x;
  const int lane  = tid & 31;
  const int wv    = tid >> 5;
  const int mBase = blockIdx.y * 64;
  const int n0    = blockIdx.x * 32;

  const int ar = tid >> 1;         // A: row
  const int ak = (tid & 1) << 4;   // A: 16-half run
  const int bg = tid >> 5;         // B: gate
  const int bn = tid & 31;         // B: column within tile

  auto loadA = [&](int buf, int kt) {
    const int kg = kt * 32 + ak;   // multiple of 16 -> runs never straddle F
    const half_t* src = (IS_ENC && kg < F_DIM)
        ? xh + ((size_t)(mBase + ar) * T_LEN + (size_t)t) * F_DIM + kg
        : hprev + (size_t)(mBase + ar) * U_DIM + (kg - (IS_ENC ? F_DIM : 0));
    half_t* dst = &sA[buf][ar * 32 + ak];
    copy16_g2l(src, dst);
    copy16_g2l(src + 8, dst + 8);
  };

  auto loadB = [&](int buf, int kt) {
    // column-major weights: 64B contiguous k-run per output column
    const half_t* src = WT + (size_t)(bg * U_DIM + n0 + bn) * K + kt * 32;
    if (kt + 1 < KT) __builtin_prefetch((const void*)(src + 32), 0, 1);
    half_t* dst = &sB[buf][bg][bn * 40];
#pragma unroll
    for (int j = 0; j < 4; ++j) copy16_g2l(src + j * 8, dst + j * 8);
  };

  v8f acc[4][2] = {};

  loadA(0, 0);
  loadB(0, 0);
  stage_fence();

#pragma unroll 2
  for (int kt = 0; kt < KT; ++kt) {
    const int buf = kt & 1;
    if (kt + 1 < KT) { loadA(buf ^ 1, kt + 1); loadB(buf ^ 1, kt + 1); }

    const int nl = lane & 15;
    const int hi = lane >> 4;
    // A fragment (ISA 7.12.2, 16-bit A 16x32): two contiguous 16B runs per lane
    const half_t* aB = &sA[buf][(wv * 16 + nl) * 32];
    v16h afrag = cat16(*(const v8h*)(aB + hi * 16), *(const v8h*)(aB + hi * 16 + 8));
    const int kof = hi * 16;
#pragma unroll
    for (int g = 0; g < 4; ++g) {
#pragma unroll
      for (int nt = 0; nt < 2; ++nt) {
        const half_t* bp = &sB[buf][g][(nt * 16 + nl) * 40 + kof];
        v16h bfrag = cat16(*(const v8h*)bp, *(const v8h*)(bp + 8));
        acc[g][nt] = __builtin_amdgcn_wmma_f32_16x16x32_f16(
            false, afrag, false, bfrag, (short)0, acc[g][nt], false, false);
      }
    }
    stage_fence();
  }

  // Gate math epilogue. C/D layout: lane n = lane%16; VGPR r -> m = r + (lane>=16)*8.
  const int mloc = mBase + wv * 16 + ((lane >> 4) << 3);
  const int nl   = lane & 15;
#pragma unroll
  for (int nt = 0; nt < 2; ++nt) {
    const int u    = n0 + nt * 16 + nl;
    const float bi  = bias[u];
    const float bfv = bias[U_DIM + u];
    const float bgv = bias[2 * U_DIM + u];
    const float bo  = bias[3 * U_DIM + u];
#pragma unroll
    for (int r = 0; r < 8; ++r) {
      const size_t idx = (size_t)(mloc + r) * U_DIM + u;
      const float si = sigmoidf_(acc[0][nt][r] + bi);
      const float sf = sigmoidf_(acc[1][nt][r] + bfv);
      const float tg = tanhf_(acc[2][nt][r] + bgv);
      const float so = sigmoidf_(acc[3][nt][r] + bo);
      const float cn = sf * cbuf[idx] + si * tg;
      cbuf[idx]  = cn;
      hnext[idx] = (half_t)(so * tanhf_(cn));
    }
  }
}

// ---------------------------------------------------------------------------
// GEMM + bias (+ optional relu): out = act(A @ WT^T + b). WT column-major [N][K].
// Grid (N/64, 256/64), block 128. WG tile 64x64; each wave 16x64.
// ---------------------------------------------------------------------------
template <bool RELU, bool OUT_HALF>
__global__ __launch_bounds__(128) void gemm_bias_kernel(
    const half_t* __restrict__ A, const half_t* __restrict__ WT,
    const float* __restrict__ bias, half_t* __restrict__ outH,
    float* __restrict__ outF, int K, int N,
    long long outStride, long long outBase) {
  __shared__ __align__(16) half_t sA[2][64 * 32];
  __shared__ __align__(16) half_t sB[2][64 * 40];

  const int tid   = threadIdx.x;
  const int lane  = tid & 31;
  const int wv    = tid >> 5;
  const int mBase = blockIdx.y * 64;
  const int nBase = blockIdx.x * 64;
  const int KT    = K / 32;

  const int ar = tid >> 1;
  const int ak = (tid & 1) << 4;
  const int bn = tid >> 1;         // B: column 0..63
  const int bj = (tid & 1) << 1;   // B: chunk pair {0,1} or {2,3}

  auto loadA = [&](int buf, int kt) {
    const half_t* src = A + (size_t)(mBase + ar) * K + kt * 32 + ak;
    half_t* dst = &sA[buf][ar * 32 + ak];
    copy16_g2l(src, dst);
    copy16_g2l(src + 8, dst + 8);
  };
  auto loadB = [&](int buf, int kt) {
    const half_t* src = WT + (size_t)(nBase + bn) * K + kt * 32 + bj * 8;
    if (kt + 1 < KT) __builtin_prefetch((const void*)(src + 32), 0, 1);
    half_t* dst = &sB[buf][bn * 40 + bj * 8];
    copy16_g2l(src, dst);
    copy16_g2l(src + 8, dst + 8);
  };

  v8f acc[4] = {};

  loadA(0, 0);
  loadB(0, 0);
  stage_fence();

#pragma unroll 2
  for (int kt = 0; kt < KT; ++kt) {
    const int buf = kt & 1;
    if (kt + 1 < KT) { loadA(buf ^ 1, kt + 1); loadB(buf ^ 1, kt + 1); }

    const int nl = lane & 15;
    const int hi = lane >> 4;
    const half_t* aB = &sA[buf][(wv * 16 + nl) * 32];
    v16h afrag = cat16(*(const v8h*)(aB + hi * 16), *(const v8h*)(aB + hi * 16 + 8));
    const int kof = hi * 16;
#pragma unroll
    for (int nt = 0; nt < 4; ++nt) {
      const half_t* bp = &sB[buf][(nt * 16 + nl) * 40 + kof];
      v16h bfrag = cat16(*(const v8h*)bp, *(const v8h*)(bp + 8));
      acc[nt] = __builtin_amdgcn_wmma_f32_16x16x32_f16(
          false, afrag, false, bfrag, (short)0, acc[nt], false, false);
    }
    stage_fence();
  }

  const int mloc = mBase + wv * 16 + ((lane >> 4) << 3);
  const int nl   = lane & 15;
#pragma unroll
  for (int nt = 0; nt < 4; ++nt) {
    const int n   = nBase + nt * 16 + nl;
    const float b = bias[n];
#pragma unroll
    for (int r = 0; r < 8; ++r) {
      float v = acc[nt][r] + b;
      if (RELU) v = v > 0.0f ? v : 0.0f;
      const int m = mloc + r;
      if (OUT_HALF) outH[(size_t)m * N + n] = (half_t)v;
      else          outF[outBase + (size_t)m * outStride + n] = v;
    }
  }
}

// ------------------------------- prep kernels ------------------------------
__global__ void cvt_f32_f16(const float* __restrict__ s, half_t* __restrict__ d, int n) {
  int i = blockIdx.x * blockDim.x + threadIdx.x;
  if (i < n) d[i] = (half_t)s[i];
}
// dst[col][k] = concat(Wx, Wh) transposed to column-major; col<4U, k<F+U
__global__ void enc_weightsT(const float* __restrict__ Wx, const float* __restrict__ Wh,
                             half_t* __restrict__ dst) {
  const int KK = F_DIM + U_DIM, NN = 4 * U_DIM;
  int i = blockIdx.x * blockDim.x + threadIdx.x;
  if (i >= NN * KK) return;
  int col = i / KK, k = i % KK;
  float v = (k < F_DIM) ? Wx[(size_t)k * NN + col]
                        : Wh[(size_t)(k - F_DIM) * NN + col];
  dst[i] = (half_t)v;
}
// dst[col][k] = (dec_Wx + dec_Wh) transposed (x_in == h_prev in the reference scan)
__global__ void dec_weightsT(const float* __restrict__ Wx, const float* __restrict__ Wh,
                             half_t* __restrict__ dst) {
  const int KK = U_DIM, NN = 4 * U_DIM;
  int i = blockIdx.x * blockDim.x + threadIdx.x;
  if (i >= NN * KK) return;
  int col = i / KK, k = i % KK;
  dst[i] = (half_t)(Wx[(size_t)k * NN + col] + Wh[(size_t)k * NN + col]);
}
// generic f32[K][N] -> f16 dst[N][K]
__global__ void transposeT(const float* __restrict__ src, half_t* __restrict__ dst,
                           int K, int N) {
  int i = blockIdx.x * blockDim.x + threadIdx.x;
  if (i >= N * K) return;
  int col = i / K, k = i % K;
  dst[i] = (half_t)src[(size_t)k * N + col];
}
__global__ void zero_state(half_t* __restrict__ h, float* __restrict__ c, int n) {
  int i = blockIdx.x * blockDim.x + threadIdx.x;
  if (i < n) { h[i] = (half_t)0.0f; c[i] = 0.0f; }
}

// --------------------------------- launcher --------------------------------
extern "C" void kernel_launch(void* const* d_in, const int* in_sizes, int n_in,
                              void* d_out, int out_size, void* d_ws, size_t ws_size,
                              hipStream_t stream) {
  const float* x   = (const float*)d_in[0];
  const float* eWx = (const float*)d_in[1];
  const float* eWh = (const float*)d_in[2];
  const float* eb  = (const float*)d_in[3];
  const float* dWx = (const float*)d_in[4];
  const float* dWh = (const float*)d_in[5];
  const float* db  = (const float*)d_in[6];
  const float* d1W = (const float*)d_in[7];
  const float* d1b = (const float*)d_in[8];
  const float* d2W = (const float*)d_in[9];
  const float* d2b = (const float*)d_in[10];
  float* out = (float*)d_out;

  char* p = (char*)d_ws;
  auto take = [&](size_t bytes) {
    char* r = p;
    p += (bytes + 255) & ~(size_t)255;
    return r;
  };
  half_t* xh    = (half_t*)take((size_t)B_SZ * T_LEN * F_DIM * sizeof(half_t));
  half_t* WencT = (half_t*)take((size_t)4 * U_DIM * (F_DIM + U_DIM) * sizeof(half_t));
  half_t* WdecT = (half_t*)take((size_t)4 * U_DIM * U_DIM * sizeof(half_t));
  half_t* D1T   = (half_t*)take((size_t)U_DIM * U_DIM * sizeof(half_t));
  half_t* D2T   = (half_t*)take((size_t)F_DIM * U_DIM * sizeof(half_t));
  half_t* hA    = (half_t*)take((size_t)B_SZ * U_DIM * sizeof(half_t));
  half_t* hB    = (half_t*)take((size_t)B_SZ * U_DIM * sizeof(half_t));
  half_t* r1    = (half_t*)take((size_t)B_SZ * U_DIM * sizeof(half_t));
  float*  cb    = (float*)take((size_t)B_SZ * U_DIM * sizeof(float));

  auto nb = [](int n) { return (n + 255) / 256; };
  cvt_f32_f16<<<nb(B_SZ * T_LEN * F_DIM), 256, 0, stream>>>(x, xh, B_SZ * T_LEN * F_DIM);
  enc_weightsT<<<nb(4 * U_DIM * (F_DIM + U_DIM)), 256, 0, stream>>>(eWx, eWh, WencT);
  dec_weightsT<<<nb(4 * U_DIM * U_DIM), 256, 0, stream>>>(dWx, dWh, WdecT);
  transposeT<<<nb(U_DIM * U_DIM), 256, 0, stream>>>(d1W, D1T, U_DIM, U_DIM);
  transposeT<<<nb(U_DIM * F_DIM), 256, 0, stream>>>(d2W, D2T, U_DIM, F_DIM);
  zero_state<<<nb(B_SZ * U_DIM), 256, 0, stream>>>(hA, cb, B_SZ * U_DIM);

  dim3 blk(128);
  dim3 gridL(U_DIM / 32, B_SZ / 64);
  half_t* hc = hA;
  half_t* hn = hB;

  // Encoder: 512 sequential steps; keep only final (h, c).
  for (int t = 0; t < T_LEN; ++t) {
    lstm_step_kernel<true><<<gridL, blk, 0, stream>>>(xh, t, hc, cb, WencT, eb, hn);
    half_t* tmp = hc; hc = hn; hn = tmp;
  }

  // Decoder + fused per-step dense head.
  dim3 gridP1(U_DIM / 64, B_SZ / 64);
  dim3 gridP2(F_DIM / 64, B_SZ / 64);
  for (int t = 0; t < T_LEN; ++t) {
    lstm_step_kernel<false><<<gridL, blk, 0, stream>>>(xh, 0, hc, cb, WdecT, db, hn);
    half_t* tmp = hc; hc = hn; hn = tmp;
    gemm_bias_kernel<true, true><<<gridP1, blk, 0, stream>>>(
        hc, D1T, d1b, r1, nullptr, U_DIM, U_DIM, 0, 0);
    gemm_bias_kernel<false, false><<<gridP2, blk, 0, stream>>>(
        r1, D2T, d2b, nullptr, out, U_DIM, F_DIM,
        (long long)T_LEN * F_DIM, (long long)t * F_DIM);
  }
  (void)in_sizes; (void)n_in; (void)out_size; (void)ws_size;
}
